// LlamaAttention_17119739642657
// MI455X (gfx1250) — matrix operand order repaired and checked by reading, not compiled, not run on hardware
//
#include <hip/hip_runtime.h>
#include <hip/hip_bf16.h>

// ---------------- problem constants ----------------
#define N_HEADS   32
#define N_KV      8
#define HDIM      128
#define HID       4096
#define BSZ       2
#define SLEN      2048
#define MROWS     (BSZ * SLEN)                       // 4096
#define QKV_N     ((N_HEADS + 2 * N_KV) * HDIM)      // 6144

typedef __attribute__((ext_vector_type(16))) __bf16 v16bf;
typedef __attribute__((ext_vector_type(8)))  float  v8f;
typedef unsigned int u32x4 __attribute__((ext_vector_type(4)));
typedef int          i32x4 __attribute__((ext_vector_type(4)));
typedef int          i32x8 __attribute__((ext_vector_type(8)));

// f32 -> bf16 with round-to-nearest-even
static __device__ __forceinline__ __bf16 f2bf(float f) {
  union { float f; unsigned u; } a; a.f = f;
  unsigned r = a.u + 0x7FFFu + ((a.u >> 16) & 1u);
  unsigned short hs = (unsigned short)(r >> 16);
  union { unsigned short s; __bf16 b; } c; c.s = hs;
  return c.b;
}

// A-fragment (16x32 bf16, M x K): this lane's 8 elems at p[0..7] and p[16..23]
static __device__ __forceinline__ v16bf load_a_frag(const __bf16* p) {
  union { v16bf v; uint4 u[2]; } t;
  t.u[0] = *(const uint4*)(p);
  t.u[1] = *(const uint4*)(p + 16);
  return t.v;
}
// B-fragment (32x16 bf16, K x N): this lane's 8 K elems at p[0..7] and p[16..23]
static __device__ __forceinline__ v16bf load_b_frag(const __bf16* p) {
  union { v16bf v; uint4 u[2]; } t;
  t.u[0] = *(const uint4*)(p);
  t.u[1] = *(const uint4*)(p + 16);
  return t.v;
}

// ---------------- CDNA5 TDM helpers ----------------
#if __has_builtin(__builtin_amdgcn_tensor_load_to_lds)
#define HAS_TDM 1
#else
#define HAS_TDM 0
#endif

static __device__ __forceinline__ void wait_tensorcnt0() {
#if __has_builtin(__builtin_amdgcn_s_wait_tensorcnt)
  __builtin_amdgcn_s_wait_tensorcnt(0);
#else
  asm volatile("s_wait_tensorcnt 0x0" ::: "memory");
#endif
}
static __device__ __forceinline__ void wait_tensorcnt2() {
#if __has_builtin(__builtin_amdgcn_s_wait_tensorcnt)
  __builtin_amdgcn_s_wait_tensorcnt(2);
#else
  asm volatile("s_wait_tensorcnt 0x2" ::: "memory");
#endif
}

#if HAS_TDM
// 2-D TDM tile load: copies tile_d1 rows of tile_d0 bf16 elements, row stride
// stride0 (elements), from global gaddr into LDS at lds_addr (packed).
// D# per CDNA5 ISA ch.8: group0 {count, lds_addr, global_addr, type=2},
// group1 {data_size=2B, tensor dims, tile dims, dim0 stride}; groups2/3 = 0.
static __device__ __forceinline__ void tdm_load_2d(unsigned lds_addr,
                                                   unsigned long long gaddr,
                                                   unsigned tile_d0,
                                                   unsigned tile_d1,
                                                   unsigned long long stride0) {
  unsigned g0[4], g1[8], gz[4] = {0u, 0u, 0u, 0u};
  g0[0] = 1u;                                             // count=1, user mode
  g0[1] = lds_addr;                                       // LDS byte address
  g0[2] = (unsigned)(gaddr & 0xFFFFFFFFu);                // global_addr[31:0]
  g0[3] = (unsigned)((gaddr >> 32) & 0x01FFFFFFu) | (2u << 30); // [56:32] | type=2
  const unsigned td0 = 0x7FFFFFFFu, td1 = 0x7FFFFFFFu;    // no OOB clipping
  g1[0] = (1u << 16);                                     // data_size=1 -> 2 bytes
  g1[1] = (td0 & 0xFFFFu) << 16;                          // tensor_dim0[15:0]
  g1[2] = (td0 >> 16) | ((td1 & 0xFFFFu) << 16);          // dim0 hi | dim1 lo
  g1[3] = (td1 >> 16) | ((tile_d0 & 0xFFFFu) << 16);      // dim1 hi | tile_dim0
  g1[4] = (tile_d1 & 0xFFFFu);                            // tile_dim1, tile_dim2=0
  g1[5] = (unsigned)(stride0 & 0xFFFFFFFFu);              // dim0_stride[31:0]
  g1[6] = (unsigned)((stride0 >> 32) & 0xFFFFu);          // dim0_stride[47:32]
  g1[7] = 0u;                                             // dim1_stride unused (2D)
  u32x4 v0; i32x8 v1; i32x4 v2, v3;
  __builtin_memcpy(&v0, g0, sizeof(v0));
  __builtin_memcpy(&v1, g1, sizeof(v1));
  __builtin_memcpy(&v2, gz, sizeof(v2));
  __builtin_memcpy(&v3, gz, sizeof(v3));
#if __clang_major__ >= 23
  __builtin_amdgcn_tensor_load_to_lds(v0, v1, v2, v3, v1, 0);
#else
  __builtin_amdgcn_tensor_load_to_lds(v0, v1, v2, v3, 0);
#endif
}
#endif  // HAS_TDM

// ---------------- elementwise conversion kernels ----------------
__global__ void f32_to_bf16_kernel(const float* __restrict__ in,
                                   __bf16* __restrict__ out, size_t n) {
  size_t i = (size_t)blockIdx.x * blockDim.x + threadIdx.x;
  if (i < n) out[i] = f2bf(in[i]);
}

// in[K][N] f32 (row-major) -> out[N][K] bf16 (row-major)
__global__ void transpose_f32_bf16(const float* __restrict__ in,
                                   __bf16* __restrict__ out, int K, int N) {
  size_t i = (size_t)blockIdx.x * blockDim.x + threadIdx.x;
  if (i >= (size_t)K * N) return;
  size_t nn = i / (size_t)K;
  size_t kk = i % (size_t)K;
  out[i] = f2bf(in[kk * (size_t)N + nn]);
}

// ---------------- bf16 WMMA GEMM: C[M][N] f32 = A[M][K] * BT[N][K] ----------------
// block = 128 threads = 4 waves; each wave computes a 32x32 tile (2x2 WMMA frags)
__global__ void gemm_bf16_f32(const __bf16* __restrict__ A,
                              const __bf16* __restrict__ BT,
                              float* __restrict__ C,
                              int M, int N, int K) {
  const int lane = threadIdx.x & 31;
  const int wave = threadIdx.x >> 5;          // 0..3
  const int wm = wave >> 1, wn = wave & 1;
  const int lr = lane & 15, hf = lane >> 4;   // lane-in-half, half index
  const int m0 = blockIdx.y * 64 + wm * 32;
  const int n0 = blockIdx.x * 64 + wn * 32;

  v8f acc[2][2] = {};
  for (int kk = 0; kk < K; kk += 32) {
    const __bf16* a0p = A + (size_t)(m0 + lr) * K + kk + hf * 8;
    const __bf16* a1p = A + (size_t)(m0 + 16 + lr) * K + kk + hf * 8;
    const __bf16* b0p = BT + (size_t)(n0 + lr) * K + kk + hf * 8;
    const __bf16* b1p = BT + (size_t)(n0 + 16 + lr) * K + kk + hf * 8;
    // hint next k-step into cache (global_prefetch_b8)
    __builtin_prefetch(a0p + 32, 0, 3);
    __builtin_prefetch(b0p + 32, 0, 3);
    v16bf a0 = load_a_frag(a0p);
    v16bf a1 = load_a_frag(a1p);
    v16bf b0 = load_b_frag(b0p);
    v16bf b1 = load_b_frag(b1p);
    acc[0][0] = __builtin_amdgcn_wmma_f32_16x16x32_bf16(false, a0, false, b0, (short)0, acc[0][0], false, false);
    acc[0][1] = __builtin_amdgcn_wmma_f32_16x16x32_bf16(false, a0, false, b1, (short)0, acc[0][1], false, false);
    acc[1][0] = __builtin_amdgcn_wmma_f32_16x16x32_bf16(false, a1, false, b0, (short)0, acc[1][0], false, false);
    acc[1][1] = __builtin_amdgcn_wmma_f32_16x16x32_bf16(false, a1, false, b1, (short)0, acc[1][1], false, false);
  }
  // C/D layout: element (M = r + 8*hf, N = lr) in VGPR r
  #pragma unroll
  for (int sm = 0; sm < 2; ++sm)
    #pragma unroll
    for (int sn = 0; sn < 2; ++sn)
      #pragma unroll
      for (int r = 0; r < 8; ++r) {
        int m = m0 + sm * 16 + r + 8 * hf;
        int n = n0 + sn * 16 + lr;
        C[(size_t)m * N + n] = acc[sm][sn][r];
      }
}

// ---------------- RoPE + reformat kernels ----------------
__global__ void rope_q_kernel(const float* __restrict__ qkv,
                              const float* __restrict__ cosT,
                              const float* __restrict__ sinT,
                              __bf16* __restrict__ Qb) {
  size_t i = (size_t)blockIdx.x * blockDim.x + threadIdx.x;
  if (i >= (size_t)BSZ * SLEN * N_HEADS * 64) return;
  int d = (int)(i & 63);
  size_t t = i >> 6;
  int h = (int)(t % N_HEADS); t /= N_HEADS;
  int s = (int)(t % SLEN);
  int b = (int)(t / SLEN);
  const float* src = qkv + ((size_t)b * SLEN + s) * QKV_N + (size_t)h * HDIM;
  float x1 = src[d], x2 = src[d + 64];
  float c = cosT[(size_t)s * 64 + d], sn = sinT[(size_t)s * 64 + d];
  __bf16* dst = Qb + (((size_t)b * N_HEADS + h) * SLEN + s) * HDIM;
  dst[d]      = f2bf(x1 * c - x2 * sn);
  dst[d + 64] = f2bf(x2 * c + x1 * sn);
}

__global__ void rope_k_kernel(const float* __restrict__ qkv,
                              const float* __restrict__ cosT,
                              const float* __restrict__ sinT,
                              __bf16* __restrict__ Kb) {
  size_t i = (size_t)blockIdx.x * blockDim.x + threadIdx.x;
  if (i >= (size_t)BSZ * SLEN * N_KV * 64) return;
  int d = (int)(i & 63);
  size_t t = i >> 6;
  int kh = (int)(t % N_KV); t /= N_KV;
  int s = (int)(t % SLEN);
  int b = (int)(t / SLEN);
  const float* src = qkv + ((size_t)b * SLEN + s) * QKV_N + (size_t)N_HEADS * HDIM + (size_t)kh * HDIM;
  float x1 = src[d], x2 = src[d + 64];
  float c = cosT[(size_t)s * 64 + d], sn = sinT[(size_t)s * 64 + d];
  __bf16* dst = Kb + (((size_t)b * N_KV + kh) * SLEN + s) * HDIM;
  dst[d]      = f2bf(x1 * c - x2 * sn);
  dst[d + 64] = f2bf(x2 * c + x1 * sn);
}

// V: qkv[row][5120 + kh*128 + d] -> VT[b][kh][d][s] bf16 (transposed for PV B-frags)
__global__ void v_transpose_kernel(const float* __restrict__ qkv,
                                   __bf16* __restrict__ VT) {
  size_t i = (size_t)blockIdx.x * blockDim.x + threadIdx.x;
  if (i >= (size_t)BSZ * SLEN * N_KV * HDIM) return;
  int d = (int)(i & 127);
  size_t t = i >> 7;
  int kh = (int)(t % N_KV); t /= N_KV;
  int s = (int)(t % SLEN);
  int b = (int)(t / SLEN);
  float v = qkv[((size_t)b * SLEN + s) * QKV_N + (size_t)(N_HEADS + N_KV) * HDIM + (size_t)kh * HDIM + d];
  VT[(((size_t)b * N_KV + kh) * HDIM + d) * SLEN + s] = f2bf(v);
}

// ---------------- flash attention: one wave per (b, h, 16-query tile) ----------------
// K and V tiles are staged into double-buffered LDS by the Tensor Data Mover
// (tensor_load_to_lds) one iteration ahead of consumption; steady-state waits are
// s_wait_tensorcnt 2 so the next tile's DMA overlaps the current tile's WMMAs.
__global__ void flash_attn_kernel(const __bf16* __restrict__ Q,
                                  const __bf16* __restrict__ Kb,
                                  const __bf16* __restrict__ VT,
                                  __bf16* __restrict__ attn) {
  const int lane = threadIdx.x;
  const int lr = lane & 15, hf = lane >> 4;
  const int q0 = blockIdx.x * 16;
  const int h  = blockIdx.y;
  const int b  = blockIdx.z;
  const int kvh = h >> 2;                      // n_rep = 32/8 = 4

  __shared__ __bf16 Klds[2][32 * HDIM];        // 2 x 8 KB: 32 keys x 128 dims
  __shared__ __bf16 Vlds[2][HDIM * 32];        // 2 x 8 KB: 128 dims x 32 keys
  __shared__ __bf16 Plds[16 * 32];             // 1 KB: probabilities tile

  const __bf16* Qp = Q  + (((size_t)b * N_HEADS + h)  * SLEN + q0) * HDIM;
  const __bf16* Kp = Kb + (((size_t)b * N_KV + kvh)   * SLEN) * HDIM;
  const __bf16* Vp = VT + (((size_t)b * N_KV + kvh)   * HDIM) * SLEN;

  // Q tile A-fragments (16 rows x 128 dims), in registers for the whole pass
  v16bf qf[4];
  #pragma unroll
  for (int kc = 0; kc < 4; ++kc)
    qf[kc] = load_a_frag(Qp + (size_t)lr * HDIM + kc * 32 + hf * 8);

  v8f accO[8] = {};                            // 16 x 128 f32 output accumulator
  float rm[8], rl[8], fac[8];
  #pragma unroll
  for (int r = 0; r < 8; ++r) { rm[r] = -1e30f; rl[r] = 0.f; }

  const float scale = 0.08838834764831845f;    // 1/sqrt(128)
  const int qHi = q0 + 15;
  const int nTiles = qHi / 32 + 1;             // key tiles of 32 (causal bound)

#if HAS_TDM
  // prologue: stage tile 0 into buffer 0
  tdm_load_2d((unsigned)(size_t)(void*)&Klds[0][0],
              (unsigned long long)(size_t)Kp,
              /*tile_d0=*/32u * HDIM, /*tile_d1=*/1u, /*stride0=*/32u * HDIM);
  tdm_load_2d((unsigned)(size_t)(void*)&Vlds[0][0],
              (unsigned long long)(size_t)Vp,
              /*tile_d0=*/32u, /*tile_d1=*/HDIM, /*stride0=*/SLEN);
#endif

  for (int ti = 0; ti < nTiles; ++ti) {
    const int j0  = ti * 32;
    const int cur = ti & 1;
#if HAS_TDM
    if (ti + 1 < nTiles) {
      const int jn = j0 + 32;
      tdm_load_2d((unsigned)(size_t)(void*)&Klds[cur ^ 1][0],
                  (unsigned long long)(size_t)(Kp + (size_t)jn * HDIM),
                  32u * HDIM, 1u, 32u * HDIM);
      tdm_load_2d((unsigned)(size_t)(void*)&Vlds[cur ^ 1][0],
                  (unsigned long long)(size_t)(Vp + jn),
                  32u, HDIM, SLEN);
      wait_tensorcnt2();                       // current pair done; next pair in flight
    } else {
      wait_tensorcnt0();
    }
#else
    // fallback: cooperative synchronous staging
    for (int idx = lane; idx < 32 * HDIM / 8; idx += 32)
      ((uint4*)&Klds[cur][0])[idx] = ((const uint4*)(Kp + (size_t)j0 * HDIM))[idx];
    for (int idx = lane; idx < HDIM * 32 / 8; idx += 32) {
      int d = idx >> 2, c = idx & 3;
      ((uint4*)&Vlds[cur][0])[idx] = *(const uint4*)(Vp + (size_t)d * SLEN + j0 + c * 8);
    }
    __syncthreads();
#endif

    // scores: two 16x16 f32 tiles over the 128-dim reduction (4 WMMA each)
    v8f sc[2] = {};
    #pragma unroll
    for (int sn = 0; sn < 2; ++sn)
      #pragma unroll
      for (int kc = 0; kc < 4; ++kc) {
        v16bf kf = load_b_frag((const __bf16*)&Klds[cur][0] + (size_t)(sn * 16 + lr) * HDIM + kc * 32 + hf * 8);
        sc[sn] = __builtin_amdgcn_wmma_f32_16x16x32_bf16(false, qf[kc], false, kf, (short)0, sc[sn], false, false);
      }

    // scale + causal mask + online softmax (row = r + 8*hf; col = j0 + sn*16 + lr)
    #pragma unroll
    for (int r = 0; r < 8; ++r) {
      int m = q0 + r + 8 * hf;
      float s0 = sc[0][r] * scale;
      float s1 = sc[1][r] * scale;
      if (j0 + lr      > m) s0 = -1e30f;
      if (j0 + 16 + lr > m) s1 = -1e30f;
      float mx = fmaxf(s0, s1);
      mx = fmaxf(mx, __shfl_xor(mx, 1, 32));
      mx = fmaxf(mx, __shfl_xor(mx, 2, 32));
      mx = fmaxf(mx, __shfl_xor(mx, 4, 32));
      mx = fmaxf(mx, __shfl_xor(mx, 8, 32));
      float mnew = fmaxf(rm[r], mx);
      float f    = __expf(rm[r] - mnew);
      float p0   = __expf(s0 - mnew);
      float p1   = __expf(s1 - mnew);
      float ps   = p0 + p1;
      ps += __shfl_xor(ps, 1, 32);
      ps += __shfl_xor(ps, 2, 32);
      ps += __shfl_xor(ps, 4, 32);
      ps += __shfl_xor(ps, 8, 32);
      rl[r] = rl[r] * f + ps;
      rm[r] = mnew;
      fac[r] = f;
      // stage P (D-layout) into LDS row-major 16x32 for A-layout reload
      Plds[(r + 8 * hf) * 32 + lr]      = f2bf(p0);
      Plds[(r + 8 * hf) * 32 + 16 + lr] = f2bf(p1);
    }
    #pragma unroll
    for (int t = 0; t < 8; ++t)
      #pragma unroll
      for (int r = 0; r < 8; ++r)
        accO[t][r] *= fac[r];
    __syncthreads();

    // P (16x32) as A-fragment; V rows (d-major) as B-fragments; O += P*V
    v16bf pf = load_a_frag((const __bf16*)Plds + (size_t)lr * 32 + hf * 8);
    #pragma unroll
    for (int t = 0; t < 8; ++t) {
      v16bf vf = load_b_frag((const __bf16*)&Vlds[cur][0] + (size_t)(t * 16 + lr) * 32 + hf * 8);
      accO[t] = __builtin_amdgcn_wmma_f32_16x16x32_bf16(false, pf, false, vf, (short)0, accO[t], false, false);
    }
    __syncthreads();
  }

  // normalize and write attn[b][s][h*128+d] bf16 (A matrix of output projection)
  #pragma unroll
  for (int r = 0; r < 8; ++r) rl[r] = 1.0f / rl[r];
  #pragma unroll
  for (int t = 0; t < 8; ++t)
    #pragma unroll
    for (int r = 0; r < 8; ++r) {
      int m = q0 + r + 8 * hf;
      int d = t * 16 + lr;
      attn[((size_t)b * SLEN + m) * (N_HEADS * HDIM) + (size_t)h * HDIM + d] =
          f2bf(accO[t][r] * rl[r]);
    }
}

// ---------------- launch ----------------
extern "C" void kernel_launch(void* const* d_in, const int* in_sizes, int n_in,
                              void* d_out, int out_size, void* d_ws, size_t ws_size,
                              hipStream_t stream) {
  const float* hidden = (const float*)d_in[0];
  const float* Wqkv   = (const float*)d_in[1];
  const float* Wo     = (const float*)d_in[2];
  const float* cosT   = (const float*)d_in[3];
  const float* sinT   = (const float*)d_in[4];
  // d_in[5] attention_mask: causal mask is computed analytically in-kernel

  char* ws = (char*)d_ws;
  size_t off = 0;
  __bf16* hiddenB = (__bf16*)(ws + off); off += (size_t)MROWS * HID * 2;        // 32 MB
  __bf16* WqkvT   = (__bf16*)(ws + off); off += (size_t)QKV_N * HID * 2;        // 48 MB
  __bf16* WoT     = (__bf16*)(ws + off); off += (size_t)HID * HID * 2;          // 32 MB
  float*  qkv     = (float*) (ws + off); off += (size_t)MROWS * QKV_N * 4;      // 96 MB
  __bf16* Qb      = (__bf16*)(ws + off); off += (size_t)BSZ * N_HEADS * SLEN * HDIM * 2; // 32 MB
  __bf16* Kb      = (__bf16*)(ws + off); off += (size_t)BSZ * N_KV * SLEN * HDIM * 2;    // 8 MB
  __bf16* VT      = (__bf16*)(ws + off); off += (size_t)BSZ * N_KV * HDIM * SLEN * 2;    // 8 MB
  __bf16* attn    = (__bf16*)(ws + off); off += (size_t)MROWS * HID * 2;        // 32 MB
  (void)in_sizes; (void)n_in; (void)out_size; (void)ws_size;

  {
    size_t n = (size_t)MROWS * HID;
    f32_to_bf16_kernel<<<dim3((unsigned)((n + 255) / 256)), 256, 0, stream>>>(hidden, hiddenB, n);
  }
  {
    size_t n = (size_t)HID * QKV_N;
    transpose_f32_bf16<<<dim3((unsigned)((n + 255) / 256)), 256, 0, stream>>>(Wqkv, WqkvT, HID, QKV_N);
  }
  {
    size_t n = (size_t)HID * HID;
    transpose_f32_bf16<<<dim3((unsigned)((n + 255) / 256)), 256, 0, stream>>>(Wo, WoT, HID, HID);
  }
  // qkv = hidden @ Wqkv  (M=4096, N=6144, K=4096)
  gemm_bf16_f32<<<dim3(QKV_N / 64, MROWS / 64), 128, 0, stream>>>(hiddenB, WqkvT, qkv, MROWS, QKV_N, HID);
  {
    size_t n = (size_t)BSZ * SLEN * N_HEADS * 64;
    rope_q_kernel<<<dim3((unsigned)((n + 255) / 256)), 256, 0, stream>>>(qkv, cosT, sinT, Qb);
  }
  {
    size_t n = (size_t)BSZ * SLEN * N_KV * 64;
    rope_k_kernel<<<dim3((unsigned)((n + 255) / 256)), 256, 0, stream>>>(qkv, cosT, sinT, Kb);
  }
  {
    size_t n = (size_t)BSZ * SLEN * N_KV * HDIM;
    v_transpose_kernel<<<dim3((unsigned)((n + 255) / 256)), 256, 0, stream>>>(qkv, VT);
  }
  // attention (flash, causal) — TDM double-buffered
  flash_attn_kernel<<<dim3(SLEN / 16, N_HEADS, BSZ), 32, 0, stream>>>(Qb, Kb, VT, attn);
  // out = attn @ Wo  (M=4096, N=4096, K=4096)
  gemm_bf16_f32<<<dim3(HID / 64, MROWS / 64), 128, 0, stream>>>(attn, WoT, (float*)d_out, MROWS, HID, HID);
}
